// KeyAttention_5557687681282
// MI455X (gfx1250) — matrix operand order repaired and checked by reading, not compiled, not run on hardware
//
#include <hip/hip_runtime.h>
#include <cmath>

#define BB 128
#define NAq 1024
#define NKq 10
#define EE 768
#define EPSq 1e-7f

typedef __attribute__((ext_vector_type(16))) __bf16 v16bf;
typedef __attribute__((ext_vector_type(8)))  __bf16 v8bf;
typedef __attribute__((ext_vector_type(8)))  float  v8f;

// ---------------------------------------------------------------------------
// Kernel 1: Z[b,k,a] = mk[k]*ma[a] * <key_k, ans_a>  via v_wmma_f32_16x16x32_bf16
// Also computes norm_key[b,k] and norm_ans[b,a] (of the masked rows).
// One block (8 wave32) per batch. key (padded 10->16 rows) lives in LDS as bf16.
// ---------------------------------------------------------------------------
__global__ __launch_bounds__(256) void k_scores(
    const float* __restrict__ ans, const float* __restrict__ mask_ans,
    const float* __restrict__ key, const float* __restrict__ mask_key,
    float* __restrict__ Zout, float* __restrict__ norm_ans,
    float* __restrict__ norm_key)
{
  __shared__ __bf16 keyS[16 * EE];     // 24 KB
  __shared__ float  ssqk[16];
  const int b   = blockIdx.x;
  const int tid = threadIdx.x;

  if (tid < 16) ssqk[tid] = 0.f;
  for (int i = tid; i < 6 * EE; i += 256) keyS[10 * EE + i] = (__bf16)0.f; // zero pad rows 10..15
  __syncthreads();
  for (int i = tid; i < NKq * EE; i += 256) {
    int k   = i / EE;
    float mk = mask_key[b * NKq + k];
    float v  = key[(size_t)(b * NKq + k) * EE + (i % EE)] * mk;
    keyS[i] = (__bf16)v;
    atomicAdd(&ssqk[k], v * v);
  }
  __syncthreads();
  if (tid < 16) {
    float s = (tid < NKq) ? ssqk[tid] : 0.f;
    norm_key[b * 16 + tid] = sqrtf(fmaxf(s, EPSq));
  }

  const int lane = tid & 31;
  const int wave = tid >> 5;
  const int m16  = lane & 15;   // A: row M / B: column N / C: column N
  const int hi   = lane >> 4;   // selects K-half for A/B, M+8 for C

  for (int t = wave; t < 64; t += 8) {          // 64 N-tiles of 16 ans rows
    const int arow = t * 16 + m16;
    const float* __restrict__ ap = ans + ((size_t)b * NAq + arow) * EE + hi * 16;
    // lead the HBM stream: prefetch this lane's row for the next tile
    if (t + 8 < 64)
      __builtin_prefetch(ans + ((size_t)b * NAq + arow + 128) * EE + hi * 16, 0, 1);
    v8f  c = {};
    float s0a = 0.f, s1a = 0.f, s2a = 0.f, s3a = 0.f;   // 4 independent ssq chains
    #pragma unroll 4
    for (int e = 0; e < EE; e += 32) {
      // A fragment (key) from LDS: lane holds K = e+16*hi+0..15
      const v8bf* p0 = (const v8bf*)(keyS + m16 * EE + e + 16 * hi);
      const v8bf* p1 = (const v8bf*)(keyS + m16 * EE + e + 16 * hi + 8);
      v8bf lo = *p0, hj = *p1;
      v16bf A;
      #pragma unroll
      for (int i = 0; i < 8; ++i) { A[i] = lo[i]; A[i + 8] = hj[i]; }
      // B fragment (ans^T): lane reads 16 contiguous f32 of its ans row
      const float4* gp = (const float4*)(ap + e);
      float4 f0 = gp[0], f1 = gp[1], f2 = gp[2], f3 = gp[3];
      s0a = fmaf(f0.x,f0.x, fmaf(f0.y,f0.y, fmaf(f0.z,f0.z, fmaf(f0.w,f0.w, s0a))));
      s1a = fmaf(f1.x,f1.x, fmaf(f1.y,f1.y, fmaf(f1.z,f1.z, fmaf(f1.w,f1.w, s1a))));
      s2a = fmaf(f2.x,f2.x, fmaf(f2.y,f2.y, fmaf(f2.z,f2.z, fmaf(f2.w,f2.w, s2a))));
      s3a = fmaf(f3.x,f3.x, fmaf(f3.y,f3.y, fmaf(f3.z,f3.z, fmaf(f3.w,f3.w, s3a))));
      v16bf Bv;
      Bv[0]=(__bf16)f0.x; Bv[1]=(__bf16)f0.y; Bv[2]=(__bf16)f0.z; Bv[3]=(__bf16)f0.w;
      Bv[4]=(__bf16)f1.x; Bv[5]=(__bf16)f1.y; Bv[6]=(__bf16)f1.z; Bv[7]=(__bf16)f1.w;
      Bv[8]=(__bf16)f2.x; Bv[9]=(__bf16)f2.y; Bv[10]=(__bf16)f2.z; Bv[11]=(__bf16)f2.w;
      Bv[12]=(__bf16)f3.x; Bv[13]=(__bf16)f3.y; Bv[14]=(__bf16)f3.z; Bv[15]=(__bf16)f3.w;
      c = __builtin_amdgcn_wmma_f32_16x16x32_bf16(false, A, false, Bv,
                                                  (short)0, c, false, false);
    }
    float ssq = (s0a + s1a) + (s2a + s3a);
    // lane pair (l, l+16) together saw every column of ans row `arow`
    ssq += __shfl_xor(ssq, 16);
    float ma = mask_ans[b * NAq + arow];
    if (hi == 0) norm_ans[b * NAq + arow] = sqrtf(fmaxf(ma * ssq, EPSq));
    #pragma unroll
    for (int r = 0; r < 8; ++r) {               // C row M = r + 8*hi
      int kk = r + 8 * hi;
      if (kk < NKq)
        Zout[((size_t)(b * NKq + kk)) * NAq + arow] = c[r] * ma;
    }
  }
}

// ---------------------------------------------------------------------------
// Kernel 2: per (b,k): softmax over NA (with mask_ans bias) and beta_key.
// ---------------------------------------------------------------------------
__global__ __launch_bounds__(256) void k_soft_ans(
    const float* __restrict__ Z, const float* __restrict__ mask_ans,
    const float* __restrict__ norm_ans, const float* __restrict__ norm_key,
    float* __restrict__ Psa, float* __restrict__ beta_key_out)
{
  __shared__ float red[256];
  const int b = blockIdx.x / NKq;
  const int k = blockIdx.x % NKq;
  const int tid = threadIdx.x;
  const float* zr = Z + (size_t)(b * NKq + k) * NAq;

  float z[4], bias[4], na[4];
  #pragma unroll
  for (int j = 0; j < 4; ++j) {
    int a   = tid + 256 * j;
    float ma = mask_ans[b * NAq + a];
    bias[j] = fabsf(ma - 1.f) * -10000.f;
    z[j]    = zr[a];
    na[j]   = norm_ans[b * NAq + a];
  }
  float m = -1e30f;
  #pragma unroll
  for (int j = 0; j < 4; ++j) m = fmaxf(m, z[j] + bias[j]);
  red[tid] = m; __syncthreads();
  for (int s = 128; s > 0; s >>= 1) { if (tid < s) red[tid] = fmaxf(red[tid], red[tid + s]); __syncthreads(); }
  m = red[0]; __syncthreads();

  float p[4], sum = 0.f;
  #pragma unroll
  for (int j = 0; j < 4; ++j) { p[j] = __expf(z[j] + bias[j] - m); sum += p[j]; }
  red[tid] = sum; __syncthreads();
  for (int s = 128; s > 0; s >>= 1) { if (tid < s) red[tid] += red[tid + s]; __syncthreads(); }
  float inv = 1.f / red[0]; __syncthreads();
  #pragma unroll
  for (int j = 0; j < 4; ++j)
    Psa[(size_t)(b * NKq + k) * NAq + tid + 256 * j] = p[j] * inv;

  // beta_key = sigmoid(5 * max_a (Z_cos + bias))
  float nk = norm_key[b * 16 + k];
  float m2 = -1e30f;
  #pragma unroll
  for (int j = 0; j < 4; ++j) m2 = fmaxf(m2, z[j] / (nk * na[j]) + bias[j]);
  red[tid] = m2; __syncthreads();
  for (int s = 128; s > 0; s >>= 1) { if (tid < s) red[tid] = fmaxf(red[tid], red[tid + s]); __syncthreads(); }
  if (tid == 0) beta_key_out[b * NKq + k] = 1.f / (1.f + __expf(-5.f * red[0]));
}

// ---------------------------------------------------------------------------
// Kernel 3: per (b,a): softmax over NK=10 (mask_key bias) and beta_ans.
// ---------------------------------------------------------------------------
__global__ __launch_bounds__(256) void k_soft_key(
    const float* __restrict__ Z, const float* __restrict__ mask_key,
    const float* __restrict__ norm_ans, const float* __restrict__ norm_key,
    float* __restrict__ Psk, float* __restrict__ beta_ans_out)
{
  const int idx = blockIdx.x * 256 + threadIdx.x;   // 0 .. B*NA-1
  const int b = idx >> 10;
  const int a = idx & 1023;
  float z[NKq], bias[NKq];
  float m = -1e30f;
  #pragma unroll
  for (int k = 0; k < NKq; ++k) {
    z[k]    = Z[(size_t)(b * NKq + k) * NAq + a];
    bias[k] = fabsf(mask_key[b * NKq + k] - 1.f) * -10000.f;
    m = fmaxf(m, z[k] + bias[k]);
  }
  float e[NKq], sum = 0.f;
  #pragma unroll
  for (int k = 0; k < NKq; ++k) { e[k] = __expf(z[k] + bias[k] - m); sum += e[k]; }
  float inv = 1.f / sum;
  #pragma unroll
  for (int k = 0; k < NKq; ++k)
    Psk[(size_t)(b * NKq + k) * NAq + a] = e[k] * inv;   // already [B,NK,NA] layout

  float na = norm_ans[b * NAq + a];
  float m2 = -1e30f;
  #pragma unroll
  for (int k = 0; k < NKq; ++k)
    m2 = fmaxf(m2, z[k] / (norm_key[b * 16 + k] * na) + bias[k]);
  beta_ans_out[b * NAq + a] = 1.f / (1.f + __expf(-5.f * m2));
}

// ---------------------------------------------------------------------------
// Kernel 4: u = max_k(U*mk*beta_key), v = max_a(V*ma*beta_ans); f = [u || v].
// V is never materialized (saves 403 MB of traffic). One block per batch.
// ---------------------------------------------------------------------------
__global__ __launch_bounds__(256) void k_uvf(
    const float* __restrict__ ans, const float* __restrict__ mask_ans,
    const float* __restrict__ key, const float* __restrict__ mask_key,
    const float* __restrict__ Psa, const float* __restrict__ Psk,
    const float* __restrict__ beta_ans, const float* __restrict__ beta_key,
    float* __restrict__ f)
{
  __shared__ float keyS[NKq][EE];   // 30 KB, masked key
  __shared__ float Pc[NKq][16];
  __shared__ float sc[16];
  const int b = blockIdx.x, tid = threadIdx.x;

  for (int i = tid; i < NKq * EE; i += 256) {
    int k = i / EE, e2 = i % EE;
    keyS[k][e2] = key[(size_t)(b * NKq + k) * EE + e2] * mask_key[b * NKq + k];
  }
  __syncthreads();

  // ---- v: stream softmax_key rows, reduce max on the fly ----
  float vmax0 = -1e30f, vmax1 = -1e30f, vmax2 = -1e30f;
  for (int a0 = 0; a0 < NAq; a0 += 16) {
    if (tid < 160) { int k = tid >> 4, aa = tid & 15;
      Pc[k][aa] = Psk[(size_t)(b * NKq + k) * NAq + a0 + aa]; }
    if (tid < 16)
      sc[tid] = mask_ans[b * NAq + a0 + tid] * beta_ans[b * NAq + a0 + tid];
    __syncthreads();
    for (int aa = 0; aa < 16; ++aa) {
      float s0 = 0.f, s1 = 0.f, s2 = 0.f;
      #pragma unroll
      for (int k = 0; k < NKq; ++k) {
        float p = Pc[k][aa];
        s0 += p * keyS[k][tid];
        s1 += p * keyS[k][tid + 256];
        s2 += p * keyS[k][tid + 512];
      }
      float g = sc[aa];
      vmax0 = fmaxf(vmax0, s0 * g);
      vmax1 = fmaxf(vmax1, s1 * g);
      vmax2 = fmaxf(vmax2, s2 * g);
    }
    __syncthreads();
  }
  f[(size_t)b * 2 * EE + EE + tid]       = vmax0;
  f[(size_t)b * 2 * EE + EE + tid + 256] = vmax1;
  f[(size_t)b * 2 * EE + EE + tid + 512] = vmax2;

  // ---- u: U[k,e] = sum_a Psa[k,a]*ma[a]*ans[a,e], second+final pass over ans ----
  float acc[NKq][3];
  #pragma unroll
  for (int k = 0; k < NKq; ++k) { acc[k][0] = 0.f; acc[k][1] = 0.f; acc[k][2] = 0.f; }
  for (int a0 = 0; a0 < NAq; a0 += 16) {
    if (tid < 160) { int k = tid >> 4, aa = tid & 15;
      Pc[k][aa] = Psa[(size_t)(b * NKq + k) * NAq + a0 + aa]
                  * mask_ans[b * NAq + a0 + aa]; }
    // prefetch next chunk of the ans stream (global_prefetch_b8)
    if (a0 + 16 < NAq)
      __builtin_prefetch(ans + ((size_t)b * NAq + a0 + 16) * EE + tid, 0, 1);
    __syncthreads();
    for (int aa = 0; aa < 16; ++aa) {
      const float* ar = ans + ((size_t)b * NAq + a0 + aa) * EE;
      float a0v = ar[tid], a1v = ar[tid + 256], a2v = ar[tid + 512];
      #pragma unroll
      for (int k = 0; k < NKq; ++k) {
        float p = Pc[k][aa];
        acc[k][0] += p * a0v; acc[k][1] += p * a1v; acc[k][2] += p * a2v;
      }
    }
    __syncthreads();
  }
  float u0 = -1e30f, u1 = -1e30f, u2 = -1e30f;
  #pragma unroll
  for (int k = 0; k < NKq; ++k) {
    float g = mask_key[b * NKq + k] * beta_key[b * NKq + k];
    u0 = fmaxf(u0, acc[k][0] * g);
    u1 = fmaxf(u1, acc[k][1] * g);
    u2 = fmaxf(u2, acc[k][2] * g);
  }
  f[(size_t)b * 2 * EE + tid]       = u0;
  f[(size_t)b * 2 * EE + tid + 256] = u1;
  f[(size_t)b * 2 * EE + tid + 512] = u2;
}

// ---------------------------------------------------------------------------
extern "C" void kernel_launch(void* const* d_in, const int* in_sizes, int n_in,
                              void* d_out, int out_size, void* d_ws, size_t ws_size,
                              hipStream_t stream) {
  const float* ans      = (const float*)d_in[0];
  const float* mask_ans = (const float*)d_in[1];
  const float* key      = (const float*)d_in[2];
  const float* mask_key = (const float*)d_in[3];

  float* out = (float*)d_out;
  // tuple: f[B,2E], Z[B,NK,NA], Psa[B,NK,NA], PskT[B,NK,NA], beta_ansT[B,1,NA], beta_keyT[B,1,NK]
  float* o_f   = out;                       // 196608
  float* o_Z   = out + 196608;              // 1310720
  float* o_Psa = out + 1507328;             // 1310720
  float* o_Psk = out + 2818048;             // 1310720
  float* o_ba  = out + 4128768;             // 131072
  float* o_bk  = out + 4259840;             // 1280

  float* w_na = (float*)d_ws;               // [B, NA]
  float* w_nk = w_na + BB * NAq;            // [B, 16]

  k_scores  <<<BB,            256, 0, stream>>>(ans, mask_ans, key, mask_key,
                                                o_Z, w_na, w_nk);
  k_soft_ans<<<BB * NKq,      256, 0, stream>>>(o_Z, mask_ans, w_na, w_nk,
                                                o_Psa, o_bk);
  k_soft_key<<<(BB * NAq)/256,256, 0, stream>>>(o_Z, mask_key, w_na, w_nk,
                                                o_Psk, o_ba);
  k_uvf     <<<BB,            256, 0, stream>>>(ans, mask_ans, key, mask_key,
                                                o_Psa, o_Psk, o_ba, o_bk, o_f);
}
